// CompressedGlobalAttention_66408784330769
// MI455X (gfx1250) — compile-verified, matrix operand order checked by reading, not run on hardware
//
#include <hip/hip_runtime.h>
#include <hip/hip_bf16.h>
#include <stdint.h>

// ---------------------------------------------------------------------------
// CompressedGlobalAttention on MI455X (gfx1250, wave32, WMMA bf16)
// B=2, S=8192, D=1024, H=16, Hd=64, R=8, local_window_start=4096 -> P=512.
// ---------------------------------------------------------------------------

typedef __attribute__((ext_vector_type(16))) __bf16 v16bf;
typedef __attribute__((ext_vector_type(8)))  __bf16 v8bf;
typedef __attribute__((ext_vector_type(8)))  float  v8f;
typedef __attribute__((ext_vector_type(4)))  int    v4i;

#define D_MODEL   1024
#define NHEADS    16
#define HEAD_DIM  64
#define SEQ       8192
#define BATCH     2
#define NPOOLS    512
#define COMP_R    8
#define MROWS     (BATCH * SEQ)     // 16384
#define KVROWS    (BATCH * NPOOLS)  // 1024

static __device__ __forceinline__ __bf16 f2bf(float f) {
  union { float f; uint32_t u; } a; a.f = f;
  uint32_t r = a.u + 0x7fffu + ((a.u >> 16) & 1u);
  union { uint16_t s; __bf16 b; } o; o.s = (uint16_t)(r >> 16);
  return o.b;
}

static __device__ __forceinline__ v16bf ldfrag(const __bf16* p0, const __bf16* p1) {
  v8bf lo = *(const v8bf*)p0;
  v8bf hi = *(const v8bf*)p1;
  v16bf r;
#pragma unroll
  for (int i = 0; i < 8; ++i) { r[i] = lo[i]; r[i + 8] = hi[i]; }
  return r;
}

static __device__ __forceinline__ v8f wmma_bf16(v16bf a, v16bf b, v8f c) {
  return __builtin_amdgcn_wmma_f32_16x16x32_bf16(false, a, false, b, (short)0, c,
                                                 false, false);
}

// 16-byte global -> LDS copy via the CDNA5 async path when the toolchain
// exposes it; plain VGPR round-trip otherwise (compile-safe either way).
static __device__ __forceinline__ void glds_cp16(__bf16* lds, const __bf16* g) {
#if __has_builtin(__builtin_amdgcn_global_load_async_to_lds_b128)
  __builtin_amdgcn_global_load_async_to_lds_b128(
      (__attribute__((address_space(1))) v4i*)g,
      (__attribute__((address_space(3))) v4i*)lds, 0, 0);
#elif __has_builtin(__builtin_amdgcn_load_to_lds)
  __builtin_amdgcn_load_to_lds((void*)g,
      (__attribute__((address_space(3))) void*)lds, 16, 0, 0);
#else
  *(v8bf*)lds = *(const v8bf*)g;
#endif
}

static __device__ __forceinline__ void glds_wait() {
#if __has_builtin(__builtin_amdgcn_global_load_async_to_lds_b128)
#if __has_builtin(__builtin_amdgcn_s_wait_asynccnt)
  __builtin_amdgcn_s_wait_asynccnt(0);
#else
  asm volatile("s_wait_asynccnt 0x0" ::: "memory");
#endif
#endif
}

// ---------------------------------------------------------------------------
// Prep kernels
// ---------------------------------------------------------------------------

__global__ __launch_bounds__(256) void cvt_x_kernel(const float* __restrict__ x,
                                                    __bf16* __restrict__ xb, int n) {
  for (int i = blockIdx.x * blockDim.x + threadIdx.x; i < n;
       i += gridDim.x * blockDim.x)
    xb[i] = f2bf(x[i]);
}

__global__ __launch_bounds__(256) void avgpool_kernel(const float* __restrict__ x,
                                                      __bf16* __restrict__ cb) {
  int idx = blockIdx.x * 256 + threadIdx.x;           // 1,048,576 threads total
  int b  = idx >> 19;
  int pl = (idx >> 10) & (NPOOLS - 1);
  int dc = idx & (D_MODEL - 1);
  const float* src = x + ((size_t)b * SEQ + (size_t)pl * COMP_R) * D_MODEL + dc;
  float s = 0.f;
#pragma unroll
  for (int r = 0; r < COMP_R; ++r) s += src[(size_t)r * D_MODEL];
  cb[idx] = f2bf(s * 0.125f);
}

// Repack W (1024x1024 fp32, row-major KxN) into WMMA B-fragment order.
__global__ __launch_bounds__(256) void pack_w_kernel(const float* __restrict__ W,
                                                     __bf16* __restrict__ out) {
  int idx = blockIdx.x * 256 + threadIdx.x;           // 1,048,576 threads total
  int kt = idx >> 15;
  int w  = idx & 32767;
  int nt = w >> 9;
  int t  = w & 511;
  int L  = t >> 4, e = t & 15;
  int hh = L >> 4;
  int n  = nt * 16 + (L & 15);
  int k  = kt * 32 + ((e < 8) ? (hh * 8 + e) : (16 + hh * 8 + (e - 8)));
  out[idx] = f2bf(W[(size_t)k * D_MODEL + n]);
}

// ---------------------------------------------------------------------------
// Generic GEMM: C(M x 1024) = A_bf16(M x 1024 row-major) @ W_packed + bias.
// modes: 0 Q->[B][H][S][64] bf16, 1 K->[B][H][P][64] bf16,
//        2 Vt->[B][H][64][P] bf16, 3 fp32 row-major.
// ---------------------------------------------------------------------------
__global__ __launch_bounds__(256) void gemm1024_kernel(
    const __bf16* __restrict__ A, const __bf16* __restrict__ Wp,
    const float* __restrict__ bias, void* __restrict__ dst, int M, int mode) {
  const int wave = threadIdx.x >> 5;
  const int lane = threadIdx.x & 31;
  const int l16  = lane & 15;
  const int hh   = lane >> 4;
  const int tilesTotal = (M >> 4) * 16;

  for (int t = blockIdx.x * 8 + wave; t < tilesTotal; t += gridDim.x * 8) {
    const int mt = t >> 4;
    const int ng = t & 15;
    v8f z = {0.f, 0.f, 0.f, 0.f, 0.f, 0.f, 0.f, 0.f};
    v8f acc[4];
#pragma unroll
    for (int j = 0; j < 4; ++j) acc[j] = z;

    const __bf16* Arow  = A + (size_t)(mt * 16 + l16) * D_MODEL;
    const __bf16* Wbase = Wp + (size_t)(ng * 4) * 512 + lane * 16;

#pragma unroll 4
    for (int kt = 0; kt < 32; ++kt) {
      v16bf af = ldfrag(Arow + kt * 32 + hh * 8, Arow + kt * 32 + 16 + hh * 8);
      const __bf16* wb = Wbase + (size_t)(kt * 64) * 512;
#pragma unroll
      for (int j = 0; j < 4; ++j) {
        v16bf bf = ldfrag(wb + j * 512, wb + j * 512 + 8);
        acc[j] = wmma_bf16(af, bf, acc[j]);
      }
    }

#pragma unroll
    for (int j = 0; j < 4; ++j) {
      const int n  = ng * 64 + j * 16 + l16;
      const float bv = bias[n];
#pragma unroll
      for (int r = 0; r < 8; ++r) {
        const int row = mt * 16 + r + 8 * hh;
        const float val = acc[j][r] + bv;
        if (mode == 0) {
          int b = row >> 13, s = row & (SEQ - 1);
          int h = n >> 6, d = n & 63;
          ((__bf16*)dst)[(((size_t)b * NHEADS + h) * SEQ + s) * HEAD_DIM + d] = f2bf(val);
        } else if (mode == 1) {
          int b = row >> 9, p = row & (NPOOLS - 1);
          int h = n >> 6, d = n & 63;
          ((__bf16*)dst)[(((size_t)b * NHEADS + h) * NPOOLS + p) * HEAD_DIM + d] = f2bf(val);
        } else if (mode == 2) {
          int b = row >> 9, p = row & (NPOOLS - 1);
          int h = n >> 6, d = n & 63;
          ((__bf16*)dst)[(((size_t)b * NHEADS + h) * HEAD_DIM + d) * NPOOLS + p] = f2bf(val);
        } else {
          ((float*)dst)[(size_t)row * D_MODEL + n] = val;
        }
      }
    }
  }
}

// ---------------------------------------------------------------------------
// Attention: block stages one head's K (512x64) and Vt (64x512) in LDS via
// async-to-LDS, then 8 waves each process 8 query tiles (every-8th tile for
// load balance) with online softmax and causal pool-chunk truncation.
// ---------------------------------------------------------------------------
__global__ __launch_bounds__(256) void attn_kernel(
    const __bf16* __restrict__ Q, const __bf16* __restrict__ K,
    const __bf16* __restrict__ Vt, __bf16* __restrict__ AO) {
  __shared__ __align__(16) __bf16 Ksh[NPOOLS * HEAD_DIM];   // 64 KB [p][d]
  __shared__ __align__(16) __bf16 Vsh[HEAD_DIM * NPOOLS];   // 64 KB [d][p]
  __shared__ __align__(16) __bf16 psh[8][16 * 32];          //  8 KB per-wave P

  const int tid  = threadIdx.x;
  const int wave = tid >> 5;
  const int lane = tid & 31;
  const int l16  = lane & 15;
  const int hh   = lane >> 4;
  const int bh   = blockIdx.x >> 3;       // 32 (b,h) pairs x 8 blocks each
  const int bsub = blockIdx.x & 7;
  const int b    = bh >> 4, h = bh & 15;

  // ---- cooperative K/V stage (async global->LDS, 16B per transfer) ----
  const __bf16* Kg = K  + (size_t)bh * NPOOLS * HEAD_DIM;
  const __bf16* Vg = Vt + (size_t)bh * HEAD_DIM * NPOOLS;
  for (int i = tid; i < (NPOOLS * HEAD_DIM) / 8; i += 256) {
    glds_cp16(&Ksh[i * 8], Kg + (size_t)i * 8);
    glds_cp16(&Vsh[i * 8], Vg + (size_t)i * 8);
  }
  glds_wait();
  __syncthreads();

  const __bf16* Qhead = Q + (size_t)bh * SEQ * HEAD_DIM;
  __bf16* sp = psh[wave];

  for (int qi = 0; qi < 8; ++qi) {
    const int t  = bsub + 8 * wave + 64 * qi;  // 0..511, strided for balance
    const int s0 = t << 4;

    // causal truncation: pools needed = floor((s0+15)/8); s0==0 keeps the
    // full loop so all-masked rows reproduce the reference's uniform softmax.
    int nchunks = 16;
    if (s0 != 0) {
      int npool = (s0 + 15) >> 3;
      nchunks = (npool + 31) >> 5;
      if (nchunks > 16) nchunks = 16;
    }

    const __bf16* qrow = Qhead + ((size_t)s0 + l16) * HEAD_DIM;
    v16bf qf0 = ldfrag(qrow + hh * 8, qrow + 16 + hh * 8);
    v16bf qf1 = ldfrag(qrow + 32 + hh * 8, qrow + 48 + hh * 8);

    v8f z = {0.f, 0.f, 0.f, 0.f, 0.f, 0.f, 0.f, 0.f};
    v8f acc[4];
#pragma unroll
    for (int j = 0; j < 4; ++j) acc[j] = z;
    float mrow[8], lrow[8];
#pragma unroll
    for (int r = 0; r < 8; ++r) { mrow[r] = -INFINITY; lrow[r] = 0.f; }

    for (int c = 0; c < nchunks; ++c) {
      const int pc = c * 32;
      // ---- scores: two 16x16 tiles from LDS-resident K ----
      v8f sA = z, sB = z;
      {
        const __bf16* k0 = Ksh + (size_t)(pc + l16) * HEAD_DIM;
        const __bf16* k1 = Ksh + (size_t)(pc + 16 + l16) * HEAD_DIM;
        v16bf kf;
        kf = ldfrag(k0 + hh * 8, k0 + 16 + hh * 8);      sA = wmma_bf16(qf0, kf, sA);
        kf = ldfrag(k1 + hh * 8, k1 + 16 + hh * 8);      sB = wmma_bf16(qf0, kf, sB);
        kf = ldfrag(k0 + 32 + hh * 8, k0 + 48 + hh * 8); sA = wmma_bf16(qf1, kf, sA);
        kf = ldfrag(k1 + 32 + hh * 8, k1 + 48 + hh * 8); sB = wmma_bf16(qf1, kf, sB);
      }

      // ---- scale + causal mask + online softmax ----
      float pA[8], pB[8], alpha[8];
#pragma unroll
      for (int r = 0; r < 8; ++r) {
        const int qpos = s0 + r + 8 * hh;
        const int p0 = pc + l16;
        const int p1 = pc + 16 + l16;
        float x0 = (qpos >= COMP_R * (p0 + 1)) ? sA[r] * 0.125f : -1e9f;
        float x1 = (qpos >= COMP_R * (p1 + 1)) ? sB[r] * 0.125f : -1e9f;
        float tmax = fmaxf(x0, x1);
        tmax = fmaxf(tmax, __shfl_xor(tmax, 1, 16));
        tmax = fmaxf(tmax, __shfl_xor(tmax, 2, 16));
        tmax = fmaxf(tmax, __shfl_xor(tmax, 4, 16));
        tmax = fmaxf(tmax, __shfl_xor(tmax, 8, 16));
        float mn = fmaxf(mrow[r], tmax);
        float a  = __expf(mrow[r] - mn);
        float e0 = __expf(x0 - mn);
        float e1 = __expf(x1 - mn);
        float ts = e0 + e1;
        ts += __shfl_xor(ts, 1, 16);
        ts += __shfl_xor(ts, 2, 16);
        ts += __shfl_xor(ts, 4, 16);
        ts += __shfl_xor(ts, 8, 16);
        lrow[r] = lrow[r] * a + ts;
        mrow[r] = mn;
        alpha[r] = a;
        pA[r] = e0;
        pB[r] = e1;
      }
#pragma unroll
      for (int j = 0; j < 4; ++j)
#pragma unroll
        for (int r = 0; r < 8; ++r) acc[j][r] *= alpha[r];

      // ---- C-layout -> A-fragment via per-wave LDS tile; DS ops are
      //      in-order per wave, so a wavefront fence is sufficient ----
#pragma unroll
      for (int r = 0; r < 8; ++r) {
        const int m = r + 8 * hh;
        sp[m * 32 + l16]      = f2bf(pA[r]);
        sp[m * 32 + 16 + l16] = f2bf(pB[r]);
      }
      __builtin_amdgcn_fence(__ATOMIC_ACQ_REL, "wavefront");
      v16bf pf = ldfrag(sp + l16 * 32 + hh * 8, sp + l16 * 32 + 16 + hh * 8);

      // ---- O += P @ V from LDS-resident Vt ----
#pragma unroll
      for (int j = 0; j < 4; ++j) {
        const __bf16* vp = Vsh + (size_t)(j * 16 + l16) * NPOOLS + pc + hh * 8;
        v16bf vf = ldfrag(vp, vp + 16);
        acc[j] = wmma_bf16(pf, vf, acc[j]);
      }
      __builtin_amdgcn_fence(__ATOMIC_ACQ_REL, "wavefront");
    }

    // ---- epilogue: O / l -> attn-out bf16 [B][S][H*64] ----
#pragma unroll
    for (int j = 0; j < 4; ++j) {
#pragma unroll
      for (int r = 0; r < 8; ++r) {
        const int m = r + 8 * hh;
        const int s = s0 + m;
        const int d = j * 16 + l16;
        float v = acc[j][r] / lrow[r];
        AO[((size_t)b * SEQ + s) * D_MODEL + h * HEAD_DIM + d] = f2bf(v);
      }
    }
  }
}

// ---------------------------------------------------------------------------
// Launch
// ---------------------------------------------------------------------------
extern "C" void kernel_launch(void* const* d_in, const int* in_sizes, int n_in,
                              void* d_out, int out_size, void* d_ws, size_t ws_size,
                              hipStream_t stream) {
  (void)in_sizes; (void)n_in; (void)out_size; (void)ws_size;
  const float* x  = (const float*)d_in[0];
  // d_in[1] = local_window_start (4096 in setup; pipeline specialized to P=512)
  const float* Wq = (const float*)d_in[2];
  const float* bq = (const float*)d_in[3];
  const float* Wk = (const float*)d_in[4];
  const float* bk = (const float*)d_in[5];
  const float* Wv = (const float*)d_in[6];
  const float* bv = (const float*)d_in[7];
  const float* Wo = (const float*)d_in[8];
  const float* bo = (const float*)d_in[9];

  char* ws = (char*)d_ws;
  __bf16* xb  = (__bf16*)(ws);                 // 32 MB: x in bf16
  __bf16* cb  = (__bf16*)(ws + 33554432);      //  2 MB: compressed bf16
  __bf16* wqp = (__bf16*)(ws + 35651584);      //  2 MB: Wq packed
  __bf16* wkp = (__bf16*)(ws + 37748736);      //  2 MB: Wk packed
  __bf16* wvp = (__bf16*)(ws + 39845888);      //  2 MB: Wv packed
  __bf16* wop = (__bf16*)(ws + 41943040);      //  2 MB: Wo packed
  __bf16* Qb  = (__bf16*)(ws + 44040192);      // 32 MB: Q  [B][H][S][64]
  __bf16* Kb  = (__bf16*)(ws + 77594624);      //  2 MB: K  [B][H][P][64]
  __bf16* Vtb = (__bf16*)(ws + 79691776);      //  2 MB: Vt [B][H][64][P]
  __bf16* AOb = (__bf16*)(ws + 81788928);      // 32 MB: attn out [B][S][1024]

  cvt_x_kernel<<<8192, 256, 0, stream>>>(x, xb, MROWS * D_MODEL);
  avgpool_kernel<<<4096, 256, 0, stream>>>(x, cb);
  pack_w_kernel<<<4096, 256, 0, stream>>>(Wq, wqp);
  pack_w_kernel<<<4096, 256, 0, stream>>>(Wk, wkp);
  pack_w_kernel<<<4096, 256, 0, stream>>>(Wv, wvp);
  pack_w_kernel<<<4096, 256, 0, stream>>>(Wo, wop);

  gemm1024_kernel<<<2048, 256, 0, stream>>>(xb, wqp, bq, (void*)Qb, MROWS, 0);
  gemm1024_kernel<<<128,  256, 0, stream>>>(cb, wkp, bk, (void*)Kb, KVROWS, 1);
  gemm1024_kernel<<<128,  256, 0, stream>>>(cb, wvp, bv, (void*)Vtb, KVROWS, 2);

  attn_kernel<<<256, 256, 0, stream>>>(Qb, Kb, Vtb, AOb);

  gemm1024_kernel<<<2048, 256, 0, stream>>>(AOb, wop, bo, d_out, MROWS, 3);
}
